// Decoder_77979426226890
// MI455X (gfx1250) — compile-verified, matrix operand order checked by reading, not run on hardware
//
#include <hip/hip_runtime.h>

// ---------------------------------------------------------------------------
// Transformer decoder (L=4, S=2048, D=512, H=8, HD=64, DFF=2048, V=50257)
// for gfx1250.  All matmuls run on v_wmma_f32_16x16x32_bf16 (f32 accumulate).
// GEMM: block = 128(M) x 64(N), 8 waves of 16x64; the 32x64 B chunk per
// K-step is staged once per block into LDS (transposed, bf16) so WMMA B
// fragments are two ds_load_b128 each.  attn_out = batched GEMM (gridDim.y=H).
// fp32 softmax / layernorm / stored attention weights.
// ---------------------------------------------------------------------------

namespace {
constexpr int kS   = 2048;
constexpr int kSE  = 2048;
constexpr int kD   = 512;
constexpr int kH   = 8;
constexpr int kHD  = 64;
constexpr int kDFF = 2048;
constexpr int kV   = 50257;
constexpr int kL   = 4;
constexpr float kEps  = 1e-6f;
constexpr float kMask = -100000.0f;
}

typedef __bf16 bf16;
typedef __attribute__((ext_vector_type(16))) __bf16 v16bf;
typedef __attribute__((ext_vector_type(8)))  __bf16 v8bf;
typedef __attribute__((ext_vector_type(8)))  float  v8f;

// Pack 16 contiguous fp32 (two groups of 8, 16 apart) into the bf16 operand
// layout: elements 0..7 -> K = base+0..7, elements 8..15 -> K = base+16..23
// (lo half-wave; hi half-wave is offset by +8 via the caller's pointer).
__device__ inline void pack_contig(v16bf& a, const float* __restrict__ p) {
#pragma unroll
  for (int e = 0; e < 8; ++e) {
    a[e]     = (bf16)p[e];
    a[8 + e] = (bf16)p[16 + e];
  }
}

// B fragment from an LDS column (32 bf16, k-major): two 16-byte loads.
__device__ inline v16bf pack_lds(const bf16* __restrict__ colbase, int hi) {
  const v8bf lo = *(const v8bf*)(colbase + 8 * hi);
  const v8bf hp = *(const v8bf*)(colbase + 8 * hi + 16);
  return __builtin_shufflevector(lo, hp, 0, 1, 2, 3, 4, 5, 6, 7,
                                 8, 9, 10, 11, 12, 13, 14, 15);
}

__device__ inline v8f wmma_bf16(const v16bf& a, const v16bf& b, const v8f& c) {
  return __builtin_amdgcn_wmma_f32_16x16x32_bf16(false, a, false, b,
                                                 (short)0, c, false, false);
}

// Epilogue for one 16x16 C tile held in v8f (m = m0 + 8*hi + e, col = n).
__device__ inline void store_tile(const v8f& c, float* __restrict__ C,
                                  const float* __restrict__ bias,
                                  const float* __restrict__ resid,
                                  int m0, int hi, int n, int N, int ldc,
                                  int relu) {
  if (n >= N) return;
  const float bv = bias ? bias[n] : 0.0f;
#pragma unroll
  for (int e = 0; e < 8; ++e) {
    const int m = m0 + 8 * hi + e;
    float v = c[e] + bv;
    if (relu) v = fmaxf(v, 0.0f);
    if (resid) v += resid[(size_t)m * ldc + n];
    C[(size_t)m * ldc + n] = v;
  }
}

// ---------------------------------------------------------------------------
// Batched GEMM: C[b] = act(A[b] (MxK, lda) @ B[b] (KxN, ldb) + bias) (+resid)
// Requires M % 128 == 0, K % 32 == 0; N ragged (clamped staging + store guard).
// ---------------------------------------------------------------------------
__global__ __launch_bounds__(256) void gemm_bf16_kernel(
    const float* __restrict__ A, const float* __restrict__ B,
    const float* __restrict__ bias, const float* __restrict__ resid,
    float* __restrict__ C, int M, int N, int K,
    int lda, int ldb, int ldc, long sA, long sB, long sC, int relu)
{
  __shared__ bf16 Bs[64 * 32];                   // [col][k], 4 KB

  const int tid  = threadIdx.x;
  const int wave = tid >> 5;
  const int lane = tid & 31;
  const int r    = lane & 15;
  const int hi   = lane >> 4;

  const int tilesN = (N + 63) >> 6;
  const int bm = blockIdx.x / tilesN;
  const int bn = blockIdx.x % tilesN;
  const int m0 = bm * 128 + wave * 16;
  const int n0 = bn * 64;

  const float* Ab = A + (size_t)blockIdx.y * sA;
  const float* Bb = B + (size_t)blockIdx.y * sB;
  float*       Cb = C + (size_t)blockIdx.y * sC;

  // Staging coordinates (uniform-per-block path selection).
  const bool vec = ((ldb & 3) == 0) && (n0 + 64 <= N);
  const int vk = tid & 31;               // vec path: row in k-chunk
  const int vc = (tid >> 5) << 3;        // vec path: first of 8 columns
  const int sc = tid & 63;               // scalar path: column
  const int sk = (tid >> 6) << 3;        // scalar path: first of 8 rows
  const int gc = (n0 + sc < N) ? (n0 + sc) : (N - 1);   // clamped column

  const float* Arow = Ab + (size_t)(m0 + r) * lda + 8 * hi;

  v8f c0 = {}, c1 = {}, c2 = {}, c3 = {};

  for (int k0 = 0; k0 < K; k0 += 32) {
    // ---- stage B[k0..k0+31, n0..n0+63] -> LDS (bf16, column-major) ----
    if (vec) {
      const float* src = Bb + (size_t)(k0 + vk) * ldb + n0 + vc;
      const float4 f0 = ((const float4*)src)[0];
      const float4 f1 = ((const float4*)src)[1];
      Bs[(vc + 0) * 32 + vk] = (bf16)f0.x;
      Bs[(vc + 1) * 32 + vk] = (bf16)f0.y;
      Bs[(vc + 2) * 32 + vk] = (bf16)f0.z;
      Bs[(vc + 3) * 32 + vk] = (bf16)f0.w;
      Bs[(vc + 4) * 32 + vk] = (bf16)f1.x;
      Bs[(vc + 5) * 32 + vk] = (bf16)f1.y;
      Bs[(vc + 6) * 32 + vk] = (bf16)f1.z;
      Bs[(vc + 7) * 32 + vk] = (bf16)f1.w;
    } else {
#pragma unroll
      for (int i = 0; i < 8; ++i)
        Bs[sc * 32 + sk + i] = (bf16)Bb[(size_t)(k0 + sk + i) * ldb + gc];
    }
    __syncthreads();

    v16bf a;
    pack_contig(a, Arow + k0);
    __builtin_prefetch(Arow + k0 + 32, 0, 1);    // global_prefetch_b8

    const v16bf b0 = pack_lds(Bs + (r +  0) * 32, hi);
    const v16bf b1 = pack_lds(Bs + (r + 16) * 32, hi);
    const v16bf b2 = pack_lds(Bs + (r + 32) * 32, hi);
    const v16bf b3 = pack_lds(Bs + (r + 48) * 32, hi);
    c0 = wmma_bf16(a, b0, c0);
    c1 = wmma_bf16(a, b1, c1);
    c2 = wmma_bf16(a, b2, c2);
    c3 = wmma_bf16(a, b3, c3);
    __syncthreads();
  }

  store_tile(c0, Cb, bias, resid, m0, hi, n0 +  0 + r, N, ldc, relu);
  store_tile(c1, Cb, bias, resid, m0, hi, n0 + 16 + r, N, ldc, relu);
  store_tile(c2, Cb, bias, resid, m0, hi, n0 + 32 + r, N, ldc, relu);
  store_tile(c3, Cb, bias, resid, m0, hi, n0 + 48 + r, N, ldc, relu);
}

// ---------------------------------------------------------------------------
// Attention scores: scores[h,q,k] = (Q_h . K_h) / 8 (+ causal mask)
// Wave block: 16 queries x 64 keys. Q and K packs are both contiguous.
// ---------------------------------------------------------------------------
__global__ __launch_bounds__(256) void attn_scores_kernel(
    const float* __restrict__ Q, const float* __restrict__ Kb,
    float* __restrict__ scores, int Sq, int Sk, int causal)
{
  const int wave = threadIdx.x >> 5;
  const int lane = threadIdx.x & 31;
  const int tilesQ = Sq >> 4, tilesK = Sk >> 6;    // 64-wide key blocks
  const long t = (long)blockIdx.x * 8 + wave;
  const long per_h = (long)tilesQ * tilesK;
  if (t >= (long)kH * per_h) return;               // whole wave exits
  const int h  = (int)(t / per_h);
  const long rem = t % per_h;
  const int m0 = (int)(rem / tilesK) << 4;
  const int n0 = (int)(rem % tilesK) << 6;
  const int r  = lane & 15;
  const int hi = lane >> 4;

  v8f c0 = {}, c1 = {}, c2 = {}, c3 = {};
  const float* qrow = Q + (size_t)(m0 + r) * kD + h * kHD + 8 * hi;
  const float* kr0 = Kb + (size_t)(n0 +  0 + r) * kD + h * kHD + 8 * hi;
  const float* kr1 = Kb + (size_t)(n0 + 16 + r) * kD + h * kHD + 8 * hi;
  const float* kr2 = Kb + (size_t)(n0 + 32 + r) * kD + h * kHD + 8 * hi;
  const float* kr3 = Kb + (size_t)(n0 + 48 + r) * kD + h * kHD + 8 * hi;
#pragma unroll
  for (int k0 = 0; k0 < kHD; k0 += 32) {
    v16bf a, b0, b1, b2, b3;
    pack_contig(a, qrow + k0);
    pack_contig(b0, kr0 + k0);
    pack_contig(b1, kr1 + k0);
    pack_contig(b2, kr2 + k0);
    pack_contig(b3, kr3 + k0);
    c0 = wmma_bf16(a, b0, c0);
    c1 = wmma_bf16(a, b1, c1);
    c2 = wmma_bf16(a, b2, c2);
    c3 = wmma_bf16(a, b3, c3);
  }

  const v8f cs[4] = {c0, c1, c2, c3};
#pragma unroll
  for (int j = 0; j < 4; ++j) {
    const int n = n0 + j * 16 + r;
#pragma unroll
    for (int e = 0; e < 8; ++e) {
      const int m = m0 + 8 * hi + e;
      float v = cs[j][e] * 0.125f;               // 1/sqrt(64)
      if (causal && n > m) v += kMask;
      scores[((size_t)h * Sq + m) * Sk + n] = v;
    }
  }
}

// ---------------------------------------------------------------------------
// Row softmax (block per row, in-place safe).
// ---------------------------------------------------------------------------
__global__ __launch_bounds__(256) void softmax_rows_kernel(
    const float* __restrict__ in, float* __restrict__ out, int cols)
{
  const size_t row = blockIdx.x;
  const float* x = in + row * (size_t)cols;
  float* y = out + row * (size_t)cols;
  __shared__ float red[256];
  const int tid = threadIdx.x;

  float m = -3.4e38f;
  for (int j = tid; j < cols; j += 256) m = fmaxf(m, x[j]);
  red[tid] = m; __syncthreads();
  for (int s = 128; s > 0; s >>= 1) {
    if (tid < s) red[tid] = fmaxf(red[tid], red[tid + s]);
    __syncthreads();
  }
  m = red[0]; __syncthreads();

  float sum = 0.0f;
  for (int j = tid; j < cols; j += 256) sum += __expf(x[j] - m);
  red[tid] = sum; __syncthreads();
  for (int s = 128; s > 0; s >>= 1) {
    if (tid < s) red[tid] += red[tid + s];
    __syncthreads();
  }
  const float inv = 1.0f / red[0];
  __syncthreads();

  for (int j = tid; j < cols; j += 256) y[j] = __expf(x[j] - m) * inv;
}

// ---------------------------------------------------------------------------
// Row layernorm (block per row of kD=512).
// ---------------------------------------------------------------------------
__global__ __launch_bounds__(256) void layernorm_rows_kernel(
    const float* __restrict__ in, const float* __restrict__ g,
    const float* __restrict__ b, float* __restrict__ out)
{
  const size_t row = blockIdx.x;
  const float* x = in + row * (size_t)kD;
  float* y = out + row * (size_t)kD;
  __shared__ float rs[256], rs2[256];
  const int tid = threadIdx.x;

  float s = 0.0f, s2 = 0.0f;
  for (int j = tid; j < kD; j += 256) { float v = x[j]; s += v; s2 += v * v; }
  rs[tid] = s; rs2[tid] = s2; __syncthreads();
  for (int k = 128; k > 0; k >>= 1) {
    if (tid < k) { rs[tid] += rs[tid + k]; rs2[tid] += rs2[tid + k]; }
    __syncthreads();
  }
  const float mean = rs[0] / (float)kD;
  const float var  = rs2[0] / (float)kD - mean * mean;
  const float inv  = rsqrtf(var + kEps);
  __syncthreads();

  for (int j = tid; j < kD; j += 256)
    y[j] = (x[j] - mean) * inv * g[j] + b[j];
}

// ---------------------------------------------------------------------------
// x[s,d] = emb[tokens[s], d] + positional encoding (pos = s+1)
// ---------------------------------------------------------------------------
__global__ __launch_bounds__(256) void embed_pos_kernel(
    const int* __restrict__ tokens, const float* __restrict__ emb,
    float* __restrict__ x)
{
  const int i = blockIdx.x * 256 + threadIdx.x;
  if (i >= kS * kD) return;
  const int s = i / kD, d = i % kD;
  const float angle = __powf(10000.0f, (float)(d & ~1) / (float)kD);
  const float ang = (float)(s + 1) / angle;
  const float pe = (d & 1) ? __cosf(ang) : __sinf(ang);
  x[i] = emb[(size_t)tokens[s] * kD + d] + pe;
}

// ---------------------------------------------------------------------------
// Host orchestration
// ---------------------------------------------------------------------------
extern "C" void kernel_launch(void* const* d_in, const int* in_sizes, int n_in,
                              void* d_out, int out_size, void* d_ws, size_t ws_size,
                              hipStream_t stream) {
  (void)in_sizes; (void)n_in; (void)out_size; (void)ws_size;

  const int*   tokens = (const int*)  d_in[0];
  const float* enc    = (const float*)d_in[1];
  const float* emb    = (const float*)d_in[2];
  const float* Wq     = (const float*)d_in[3];
  const float* bq     = (const float*)d_in[4];
  const float* Wk     = (const float*)d_in[5];
  const float* bk     = (const float*)d_in[6];
  const float* Wv     = (const float*)d_in[7];
  const float* bv     = (const float*)d_in[8];
  const float* Wo     = (const float*)d_in[9];
  const float* bo     = (const float*)d_in[10];
  const float* lng    = (const float*)d_in[11];
  const float* lnb    = (const float*)d_in[12];
  const float* W1     = (const float*)d_in[13];
  const float* b1     = (const float*)d_in[14];
  const float* W2     = (const float*)d_in[15];
  const float* b2     = (const float*)d_in[16];
  const float* Wf     = (const float*)d_in[17];
  const float* bfv    = (const float*)d_in[18];

  // d_out = probs (S,V) | masked_ws[0] (H,S,S) | cross_ws[3] (H,S,SE)
  float* probs = (float*)d_out;
  float* mw0   = probs + (size_t)kS * kV;
  float* cw3   = mw0 + (size_t)kH * kS * kS;

  // Workspace: 6*S*D + S*DFF floats (~42 MB)
  const size_t SD = (size_t)kS * kD;
  float* x  = (float*)d_ws;
  float* y  = x  + SD;
  float* q  = y  + SD;
  float* kb = q  + SD;
  float* vb = kb + SD;
  float* o  = vb + SD;
  float* hb = o  + SD;    // S x DFF

  // Plain GEMM: C(MxN) = act(A(MxK) @ B(KxN) + bias) (+resid), no batching.
  auto gemm = [&](const float* A, const float* B, const float* bias,
                  const float* resid, float* C, int M, int N, int K, int relu) {
    dim3 grid((unsigned)((M / 128) * ((N + 63) / 64)), 1);
    gemm_bf16_kernel<<<grid, 256, 0, stream>>>(A, B, bias, resid, C, M, N, K,
                                               K, N, N, 0, 0, 0, relu);
  };
  // Per-head batched GEMM: O_h(Sq x 64) = W_h(Sq x Sk) @ V_h(Sk x 64)
  auto attn_out = [&](const float* Wt, const float* Vb, float* O, int Sk) {
    dim3 grid((unsigned)(kS / 128), kH);
    gemm_bf16_kernel<<<grid, 256, 0, stream>>>(
        Wt, Vb, nullptr, nullptr, O, kS, kHD, Sk,
        /*lda=*/Sk, /*ldb=*/kD, /*ldc=*/kD,
        /*sA=*/(long)kS * Sk, /*sB=*/kHD, /*sC=*/kHD, 0);
  };
  auto scores = [&](const float* Q, const float* Kb2, float* sc, int Sk,
                    int causal) {
    long tiles = (long)kH * (kS / 16) * (Sk / 64);
    attn_scores_kernel<<<(unsigned)((tiles + 7) / 8), 256, 0, stream>>>(
        Q, Kb2, sc, kS, Sk, causal);
  };

  embed_pos_kernel<<<(kS * kD + 255) / 256, 256, 0, stream>>>(tokens, emb, x);

  for (int l = 0; l < kL; ++l) {
    const size_t wi0 = (size_t)(l * 2 + 0) * kD * kD;
    const size_t wi1 = (size_t)(l * 2 + 1) * kD * kD;
    const size_t bi0 = (size_t)(l * 2 + 0) * kD;
    const size_t bi1 = (size_t)(l * 2 + 1) * kD;

    // ---- masked self-attention ----
    gemm(x, Wq + wi0, bq + bi0, nullptr, q,  kS, kD, kD, 0);
    gemm(x, Wk + wi0, bk + bi0, nullptr, kb, kS, kD, kD, 0);
    gemm(x, Wv + wi0, bv + bi0, nullptr, vb, kS, kD, kD, 0);

    float* sc = (l == 0) ? mw0 : cw3;   // layer-0 weights kept; else scratch
    scores(q, kb, sc, kS, 1);
    softmax_rows_kernel<<<kH * kS, 256, 0, stream>>>(sc, sc, kS);
    attn_out(sc, vb, o, kS);
    gemm(o, Wo + wi0, bo + bi0, x, y, kS, kD, kD, 0);
    layernorm_rows_kernel<<<kS, 256, 0, stream>>>(
        y, lng + (size_t)(l * 3 + 0) * kD, lnb + (size_t)(l * 3 + 0) * kD, x);

    // ---- cross-attention (K,V from encoder output) ----
    gemm(x,   Wq + wi1, bq + bi1, nullptr, q,  kS,  kD, kD, 0);
    gemm(enc, Wk + wi1, bk + bi1, nullptr, kb, kSE, kD, kD, 0);
    gemm(enc, Wv + wi1, bv + bi1, nullptr, vb, kSE, kD, kD, 0);
    scores(q, kb, cw3, kSE, 0);
    softmax_rows_kernel<<<kH * kS, 256, 0, stream>>>(cw3, cw3, kSE);
    attn_out(cw3, vb, o, kSE);
    gemm(o, Wo + wi1, bo + bi1, x, y, kS, kD, kD, 0);
    layernorm_rows_kernel<<<kS, 256, 0, stream>>>(
        y, lng + (size_t)(l * 3 + 1) * kD, lnb + (size_t)(l * 3 + 1) * kD, x);

    // ---- feed-forward ----
    gemm(x,  W1 + (size_t)l * kD * kDFF, b1 + (size_t)l * kDFF, nullptr, hb,
         kS, kDFF, kD, 1);
    gemm(hb, W2 + (size_t)l * kDFF * kD, b2 + (size_t)l * kD, x, y,
         kS, kD, kDFF, 0);
    layernorm_rows_kernel<<<kS, 256, 0, stream>>>(
        y, lng + (size_t)(l * 3 + 2) * kD, lnb + (size_t)(l * 3 + 2) * kD, x);
  }

  // Final projection: logits straight into the probs region, softmax in place.
  gemm(x, Wf, bfv, nullptr, probs, kS, kV, kD, 0);
  softmax_rows_kernel<<<kS, 256, 0, stream>>>(probs, probs, kV);
}
